// PairUpdateBlock_3332894622603
// MI455X (gfx1250) — matrix operand order repaired and checked by reading, not compiled, not run on hardware
//
#include <hip/hip_runtime.h>
#include <hip/hip_bf16.h>
#include <stdint.h>

#define B_  2
#define S_  512
#define SD_ 512
#define PD_ 128
#define AD_ 64
#define HH_ 256
#define P_  256

typedef __attribute__((ext_vector_type(16))) _Float16 v16h;
typedef __attribute__((ext_vector_type(8)))  float    v8f;

union Frag { v16h v; uint32_t u[8]; };

// A operand (16x32 f16): lane m = lane&15 ; VGPR v holds K = 16*(v/4) + 8*(lane/16) + 2*(v%4)+h
__device__ __forceinline__ v16h load_A(const _Float16* p, int ld, int row0, int k0, int lane) {
  Frag f;
  const int r  = row0 + (lane & 15);
  const int kb = k0 + 8 * (lane >> 4);
  #pragma unroll
  for (int v = 0; v < 8; ++v) {
    int k = kb + 16 * (v >> 2) + 2 * (v & 3);
    f.u[v] = *(const uint32_t*)(p + (size_t)r * ld + k);
  }
  return f.v;
}

// B operand (32x16 f16) from a [N][K] buffer: lane n = n0 + lane&15 ; VGPR v holds K = 16*(v/4) + 8*(lane/16) + 2*(v%4)+h
__device__ __forceinline__ v16h load_B(const _Float16* p, int ld, int n0, int k0, int lane) {
  Frag f;
  const int n  = n0 + (lane & 15);
  const int kb = k0 + 8 * (lane >> 4);
  #pragma unroll
  for (int v = 0; v < 8; ++v)
    f.u[v] = *(const uint32_t*)(p + (size_t)n * ld + kb + 16 * (v >> 2) + 2 * (v & 3));
  return f.v;
}

__device__ __forceinline__ v8f wmma_f16(v16h a, v16h b, v8f c) {
  return __builtin_amdgcn_wmma_f32_16x16x32_f16(false, a, false, b, (short)0, c, false, false);
}

__device__ float block_reduce(float v, float* red) {
  int t = threadIdx.x;
  __syncthreads();
  red[t] = v;
  __syncthreads();
  for (int s = 128; s > 0; s >>= 1) {
    if (t < s) red[t] += red[t + s];
    __syncthreads();
  }
  return red[0];
}

// ---------------- Kernel 1: seq LN + avgpool + combined projections ----------------
__global__ void k_prep(const float* __restrict__ seq, const float* __restrict__ sn_g,
                       const float* __restrict__ sn_b,
                       const float* __restrict__ Wq, const float* __restrict__ Wk,
                       const float* __restrict__ Wyq, const float* __restrict__ Wyk,
                       float* __restrict__ a_side, float* __restrict__ b_side) {
  __shared__ float xs[SD_];
  __shared__ float gx[SD_];
  __shared__ float red[256];
  const int bp = blockIdx.x;
  const int b = bp / P_, p = bp % P_;
  const int t = threadIdx.x;
  const float* r0 = seq + ((size_t)b * S_ + 2 * p) * SD_;
  const float* r1 = r0 + SD_;
  float s0 = 0.f, q0 = 0.f, s1 = 0.f, q1 = 0.f;
  for (int k = t; k < SD_; k += 256) {
    float a = r0[k], c = r1[k];
    s0 += a; q0 += a * a; s1 += c; q1 += c * c;
  }
  s0 = block_reduce(s0, red);
  q0 = block_reduce(q0, red);
  s1 = block_reduce(s1, red);
  q1 = block_reduce(q1, red);
  float mu0 = s0 / SD_, mu1 = s1 / SD_;
  float rs0 = rsqrtf(q0 / SD_ - mu0 * mu0 + 1e-5f);
  float rs1 = rsqrtf(q1 / SD_ - mu1 * mu1 + 1e-5f);
  for (int k = t; k < SD_; k += 256) {
    float a = (r0[k] - mu0) * rs0 * sn_g[k] + sn_b[k];
    float c = (r1[k] - mu1) * rs1 * sn_g[k] + sn_b[k];
    float m = 0.5f * (a + c);
    xs[k] = m;
    gx[k] = 0.5f * m * (1.0f + erff(m * 0.70710678118f));   // exact gelu
  }
  __syncthreads();
  if (t < PD_) {
    int d = t;
    float acc = 0.f;
    for (int k = 0; k < SD_; ++k)
      acc += xs[k] * Wq[(size_t)k * PD_ + d] + gx[k] * Wyq[(size_t)k * PD_ + d];
    a_side[((size_t)b * P_ + p) * PD_ + d] = acc;
  } else {
    int d = t - PD_;
    float acc = 0.f;
    for (int k = 0; k < SD_; ++k)
      acc += xs[k] * Wk[(size_t)k * PD_ + d] + gx[k] * Wyk[(size_t)k * PD_ + d];
    b_side[((size_t)b * P_ + p) * PD_ + d] = acc;
  }
}

// ---------------- Weight convert+transpose: dst[N][K] f16 = src[K][N] f32 * scale --------
__global__ void k_cvtT(const float* __restrict__ src, _Float16* __restrict__ dst,
                       int K, int N, float scale) {
  int i = blockIdx.x * 256 + threadIdx.x;
  if (i < K * N) {
    int n = i / K, k = i % K;
    dst[i] = (_Float16)(src[(size_t)k * N + n] * scale);
  }
}

// ---------------- Kernel 2: x = pair_input + pre @ Wout + bout ----------------
__global__ void k_pair(const float* __restrict__ a_side, const float* __restrict__ b_side,
                       const float* __restrict__ Wdist, const _Float16* __restrict__ WoutT,
                       const float* __restrict__ bout, const float* __restrict__ pair_in,
                       float* __restrict__ X) {
  extern __shared__ char smem[];
  _Float16* pre = (_Float16*)smem;          // [P_][PD_]  64KB
  _Float16* wB  = pre + P_ * PD_;           // [PD_][PD_] 32KB
  const int i = blockIdx.x, b = blockIdx.y;
  const int t = threadIdx.x;
  {
    const uint32_t* src = (const uint32_t*)WoutT;
    uint32_t* dst = (uint32_t*)wB;
    for (int idx = t; idx < PD_ * PD_ / 2; idx += 256) dst[idx] = src[idx];
  }
  {
    const int s = t;
    float dd  = (float)i - (float)s;
    float dsc = fabsf(dd) * (1.0f / (float)(P_ - 1));
    float sg  = (dd > 0.f) ? 1.f : ((dd < 0.f) ? -1.f : 0.f);
    float ex  = __expf(-dsc);
    const float* av = a_side + ((size_t)b * P_ + i) * PD_;
    const float* bv = b_side + ((size_t)b * P_ + s) * PD_;
    for (int dk = 0; dk < PD_; ++dk) {
      float val = av[dk] + bv[dk] + dsc * Wdist[dk] + sg * Wdist[PD_ + dk] + ex * Wdist[2 * PD_ + dk];
      pre[s * PD_ + dk] = (_Float16)val;
    }
  }
  __syncthreads();
  const int lane = t & 31, wave = t >> 5;
  const float* pi = pair_in + (((size_t)b * P_ + i) * P_) * PD_;
  float* xo = X + (((size_t)b * P_ + i) * P_) * PD_;
  #pragma unroll
  for (int mi = 0; mi < 2; ++mi) {
    const int mt = wave * 2 + mi;
    v16h a[4];                                   // register-blocked A (reused across 8 n-tiles)
    #pragma unroll
    for (int kk = 0; kk < 4; ++kk) a[kk] = load_A(pre, PD_, mt * 16, kk * 32, lane);
    #pragma unroll
    for (int nt = 0; nt < 8; ++nt) {
      v8f acc = {};
      #pragma unroll
      for (int kk = 0; kk < 4; ++kk)
        acc = wmma_f16(a[kk], load_B(wB, PD_, nt * 16, kk * 32, lane), acc);
      int n  = nt * 16 + (lane & 15);
      int mb = mt * 16 + 8 * (lane >> 4);
      float bo = bout[n];
      float resid[8];                            // batched residual loads (one clause, one wait)
      #pragma unroll
      for (int r = 0; r < 8; ++r) resid[r] = pi[(size_t)(mb + r) * PD_ + n];
      #pragma unroll
      for (int r = 0; r < 8; ++r)
        xo[(size_t)(mb + r) * PD_ + n] = acc[r] + resid[r] + bo;
    }
  }
}

// ---------------- Kernel 3: row LN + Q/K/V projections ----------------
__global__ void k_lnqkv(const float* __restrict__ X, const float* __restrict__ rn_g,
                        const float* __restrict__ rn_b, const _Float16* __restrict__ WqkvT,
                        const float* __restrict__ brv,
                        _Float16* __restrict__ QA, _Float16* __restrict__ KA,
                        _Float16* __restrict__ VA) {
  extern __shared__ char smem[];
  _Float16* yn = (_Float16*)smem;           // [P_][PD_]       64KB
  _Float16* wq = yn + P_ * PD_;             // [3][AD_][PD_]   48KB
  const int i = blockIdx.x, b = blockIdx.y;
  const int t = threadIdx.x;
  {
    const uint32_t* src = (const uint32_t*)WqkvT;
    uint32_t* dst = (uint32_t*)wq;
    for (int idx = t; idx < 3 * AD_ * PD_ / 2; idx += 256) dst[idx] = src[idx];
  }
  {
    const float* xr = X + (((size_t)b * P_ + i) * P_ + t) * PD_;
    float s = 0.f, q = 0.f;
    for (int d = 0; d < PD_; ++d) { float v = xr[d]; s += v; q += v * v; }
    float mu = s * (1.f / PD_);
    float rs = rsqrtf(q * (1.f / PD_) - mu * mu + 1e-5f);
    for (int d = 0; d < PD_; ++d)
      yn[t * PD_ + d] = (_Float16)((xr[d] - mu) * rs * rn_g[d] + rn_b[d]);
  }
  __syncthreads();
  const int lane = t & 31, wave = t >> 5;
  const size_t base = ((size_t)b * P_ + i) * P_ * AD_;
  // 48 (mat, mtile) groups over 8 waves; A fragments reused across the 4 n-tiles
  #pragma unroll
  for (int g = 0; g < 6; ++g) {
    int gid = wave * 6 + g;                  // 0..47
    int mat = gid >> 4, mt = gid & 15;
    v16h a[4];
    #pragma unroll
    for (int kk = 0; kk < 4; ++kk) a[kk] = load_A(yn, PD_, mt * 16, kk * 32, lane);
    _Float16* dst = (mat == 0) ? QA : ((mat == 1) ? KA : VA);
    const _Float16* wb = wq + mat * AD_ * PD_;
    #pragma unroll
    for (int nt = 0; nt < 4; ++nt) {
      v8f acc = {};
      #pragma unroll
      for (int kk = 0; kk < 4; ++kk)
        acc = wmma_f16(a[kk], load_B(wb, PD_, nt * 16, kk * 32, lane), acc);
      int n  = nt * 16 + (lane & 15);
      int mb = mt * 16 + 8 * (lane >> 4);
      float bias = (mat == 2) ? brv[n] : 0.f;
      #pragma unroll
      for (int r = 0; r < 8; ++r) {
        int row = mb + r;
        dst[base + (size_t)row * AD_ + n] = (_Float16)(acc[r] + bias);
      }
    }
  }
}

// ---------------- Kernel 4: flash row-attention (online softmax) ----------------
__global__ void k_attn(const _Float16* __restrict__ QA, const _Float16* __restrict__ KA,
                       const _Float16* __restrict__ VA, _Float16* __restrict__ O) {
  extern __shared__ char smem[];
  _Float16* ka  = (_Float16*)smem;          // [P_][AD_]  32KB  (B for logits: [n=s][k=d])
  _Float16* vaT = ka + P_ * AD_;            // [AD_][P_]  32KB  (B for o:      [n=d][k=s])
  _Float16* ps  = vaT + AD_ * P_;           // [8][16][32] 8KB  per-wave P scratch
  const int i = blockIdx.x, b = blockIdx.y;
  const int t = threadIdx.x;
  const size_t base = ((size_t)b * P_ + i) * P_ * AD_;
  {
    const uint32_t* src = (const uint32_t*)(KA + base);
    uint32_t* dst = (uint32_t*)ka;
    for (int idx = t; idx < P_ * AD_ / 2; idx += 256) dst[idx] = src[idx];
    const _Float16* vsrc = VA + base;
    for (int idx = t; idx < P_ * AD_; idx += 256) {
      int s = idx >> 6, d = idx & 63;
      vaT[d * P_ + s] = vsrc[idx];
    }
  }
  __syncthreads();
  const int lane = t & 31, wave = t >> 5;
  _Float16* myps = ps + wave * 16 * 32;
  for (int half = 0; half < 2; ++half) {
    int mt = wave + 8 * half;
    v16h aq0 = load_A(QA + base, AD_, mt * 16, 0, lane);   // scale folded into Wrq
    v16h aq1 = load_A(QA + base, AD_, mt * 16, 32, lane);
    v8f o0 = {}, o1 = {}, o2 = {}, o3 = {};
    float rmax[8], rsum[8];
    #pragma unroll
    for (int r = 0; r < 8; ++r) { rmax[r] = -3.0e38f; rsum[r] = 0.f; }
    for (int st = 0; st < 8; ++st) {       // 32 keys per step
      v8f s0 = {}, s1 = {};
      s0 = wmma_f16(aq0, load_B(ka, AD_, st * 32, 0, lane), s0);
      s0 = wmma_f16(aq1, load_B(ka, AD_, st * 32, 32, lane), s0);
      s1 = wmma_f16(aq0, load_B(ka, AD_, st * 32 + 16, 0, lane), s1);
      s1 = wmma_f16(aq1, load_B(ka, AD_, st * 32 + 16, 32, lane), s1);
      float tmax[8];
      #pragma unroll
      for (int r = 0; r < 8; ++r) tmax[r] = fmaxf(s0[r], s1[r]);
      #pragma unroll
      for (int off = 1; off < 16; off <<= 1) {
        #pragma unroll
        for (int r = 0; r < 8; ++r) tmax[r] = fmaxf(tmax[r], __shfl_xor(tmax[r], off, 32));
      }
      float p0[8], p1[8], tsum[8];
      #pragma unroll
      for (int r = 0; r < 8; ++r) {
        float nm  = fmaxf(rmax[r], tmax[r]);
        float cor = __expf(rmax[r] - nm);
        rmax[r] = nm;
        rsum[r] *= cor;
        o0[r] *= cor; o1[r] *= cor; o2[r] *= cor; o3[r] *= cor;
        p0[r] = __expf(s0[r] - nm);
        p1[r] = __expf(s1[r] - nm);
        tsum[r] = p0[r] + p1[r];
      }
      #pragma unroll
      for (int off = 1; off < 16; off <<= 1) {
        #pragma unroll
        for (int r = 0; r < 8; ++r) tsum[r] += __shfl_xor(tsum[r], off, 32);
      }
      #pragma unroll
      for (int r = 0; r < 8; ++r) rsum[r] += tsum[r];
      // relayout P (C layout) -> A layout via per-wave LDS scratch
      #pragma unroll
      for (int r = 0; r < 8; ++r) {
        int m = r + 8 * (lane >> 4);
        myps[m * 32 + (lane & 15)]      = (_Float16)p0[r];
        myps[m * 32 + 16 + (lane & 15)] = (_Float16)p1[r];
      }
      v16h ap = load_A(myps, 32, 0, 0, lane);
      o0 = wmma_f16(ap, load_B(vaT, P_, 0,  st * 32, lane), o0);
      o1 = wmma_f16(ap, load_B(vaT, P_, 16, st * 32, lane), o1);
      o2 = wmma_f16(ap, load_B(vaT, P_, 32, st * 32, lane), o2);
      o3 = wmma_f16(ap, load_B(vaT, P_, 48, st * 32, lane), o3);
    }
    _Float16* od = O + base;
    int mb = mt * 16 + 8 * (lane >> 4);
    #pragma unroll
    for (int r = 0; r < 8; ++r) {
      float inv = 1.0f / rsum[r];
      int row = mb + r;
      od[(size_t)row * AD_ + 0  + (lane & 15)] = (_Float16)(o0[r] * inv);
      od[(size_t)row * AD_ + 16 + (lane & 15)] = (_Float16)(o1[r] * inv);
      od[(size_t)row * AD_ + 32 + (lane & 15)] = (_Float16)(o2[r] * inv);
      od[(size_t)row * AD_ + 48 + (lane & 15)] = (_Float16)(o3[r] * inv);
    }
  }
}

// ---------------- Kernel 5: x2 = x + o @ Wro + bro ----------------
__global__ void k_attnout(const _Float16* __restrict__ O, const _Float16* __restrict__ WroT,
                          const float* __restrict__ bro, const float* __restrict__ X,
                          float* __restrict__ X2) {
  extern __shared__ char smem[];
  _Float16* o16 = (_Float16*)smem;          // [P_][AD_]  32KB
  _Float16* wB  = o16 + P_ * AD_;           // [PD_][AD_] 16KB
  const int i = blockIdx.x, b = blockIdx.y;
  const int t = threadIdx.x;
  const size_t baseA = ((size_t)b * P_ + i) * P_ * AD_;
  const size_t baseX = ((size_t)b * P_ + i) * P_ * PD_;
  {
    const uint32_t* src = (const uint32_t*)(O + baseA);
    uint32_t* dst = (uint32_t*)o16;
    for (int idx = t; idx < P_ * AD_ / 2; idx += 256) dst[idx] = src[idx];
    const uint32_t* wsrc = (const uint32_t*)WroT;
    uint32_t* wdst = (uint32_t*)wB;
    for (int idx = t; idx < PD_ * AD_ / 2; idx += 256) wdst[idx] = wsrc[idx];
  }
  __syncthreads();
  const int lane = t & 31, wave = t >> 5;
  #pragma unroll
  for (int mi = 0; mi < 2; ++mi) {
    const int mt = wave * 2 + mi;
    v16h a0 = load_A(o16, AD_, mt * 16, 0, lane);
    v16h a1 = load_A(o16, AD_, mt * 16, 32, lane);
    #pragma unroll
    for (int nt = 0; nt < 8; ++nt) {
      v8f acc = {};
      acc = wmma_f16(a0, load_B(wB, AD_, nt * 16, 0, lane),  acc);
      acc = wmma_f16(a1, load_B(wB, AD_, nt * 16, 32, lane), acc);
      int n  = nt * 16 + (lane & 15);
      int mb = mt * 16 + 8 * (lane >> 4);
      float bb = bro[n];
      float resid[8];                            // batched residual loads
      #pragma unroll
      for (int r = 0; r < 8; ++r) resid[r] = X[baseX + (size_t)(mb + r) * PD_ + n];
      #pragma unroll
      for (int r = 0; r < 8; ++r)
        X2[baseX + (size_t)(mb + r) * PD_ + n] = resid[r] + acc[r] + bb;
    }
  }
}

// ---------------- Kernel 6: h = LN(x2); g = relu(h @ W1 + b1) ----------------
__global__ void k_mlp1(const float* __restrict__ X2, const float* __restrict__ mn_g,
                       const float* __restrict__ mn_b, const _Float16* __restrict__ W1T,
                       const float* __restrict__ b1, _Float16* __restrict__ G) {
  extern __shared__ char smem[];
  _Float16* h16 = (_Float16*)smem;          // [P_][PD_]  64KB
  _Float16* wB  = h16 + P_ * PD_;           // [HH_][PD_] 64KB
  const int i = blockIdx.x, b = blockIdx.y;
  const int t = threadIdx.x;
  const size_t baseX = ((size_t)b * P_ + i) * P_ * PD_;
  const size_t baseH = ((size_t)b * P_ + i) * P_ * HH_;
  {
    const uint32_t* src = (const uint32_t*)W1T;
    uint32_t* dst = (uint32_t*)wB;
    for (int idx = t; idx < HH_ * PD_ / 2; idx += 256) dst[idx] = src[idx];
  }
  {
    const float* xr = X2 + baseX + (size_t)t * PD_;
    float s = 0.f, q = 0.f;
    for (int d = 0; d < PD_; ++d) { float v = xr[d]; s += v; q += v * v; }
    float mu = s * (1.f / PD_);
    float rs = rsqrtf(q * (1.f / PD_) - mu * mu + 1e-5f);
    for (int d = 0; d < PD_; ++d)
      h16[t * PD_ + d] = (_Float16)((xr[d] - mu) * rs * mn_g[d] + mn_b[d]);
  }
  __syncthreads();
  const int lane = t & 31, wave = t >> 5;
  #pragma unroll
  for (int mi = 0; mi < 2; ++mi) {
    const int mt = wave * 2 + mi;
    v16h a[4];                                   // reused across all 16 n-tiles
    #pragma unroll
    for (int kk = 0; kk < 4; ++kk) a[kk] = load_A(h16, PD_, mt * 16, kk * 32, lane);
    #pragma unroll
    for (int nt = 0; nt < 16; ++nt) {
      v8f acc = {};
      #pragma unroll
      for (int kk = 0; kk < 4; ++kk)
        acc = wmma_f16(a[kk], load_B(wB, PD_, nt * 16, kk * 32, lane), acc);
      int n  = nt * 16 + (lane & 15);
      int mb = mt * 16 + 8 * (lane >> 4);
      float bb = b1[n];
      #pragma unroll
      for (int r = 0; r < 8; ++r) {
        int row = mb + r;
        G[baseH + (size_t)row * HH_ + n] = (_Float16)fmaxf(acc[r] + bb, 0.f);
      }
    }
  }
}

// ---------------- Kernel 7: out = x2 + g @ W2 + b2 (in place into d_out) ----------------
__global__ void k_mlp2(const _Float16* __restrict__ G, const _Float16* __restrict__ W2T,
                       const float* __restrict__ b2, float* __restrict__ X2out) {
  extern __shared__ char smem[];
  _Float16* wB = (_Float16*)smem;           // [PD_][HH_] 64KB
  const int i = blockIdx.x, b = blockIdx.y;
  const int t = threadIdx.x;
  const size_t baseG = ((size_t)b * P_ + i) * P_ * HH_;
  const size_t baseX = ((size_t)b * P_ + i) * P_ * PD_;
  {
    const uint32_t* src = (const uint32_t*)W2T;
    uint32_t* dst = (uint32_t*)wB;
    for (int idx = t; idx < PD_ * HH_ / 2; idx += 256) dst[idx] = src[idx];
  }
  __syncthreads();
  const int lane = t & 31, wave = t >> 5;
  __builtin_prefetch(G + baseG + (size_t)t * HH_, 0, 1);    // warm L2/L0 path
  #pragma unroll
  for (int mi = 0; mi < 2; ++mi) {
    const int mt = wave * 2 + mi;
    v16h a[8];                                   // A from L2-resident global, loaded once
    #pragma unroll
    for (int kk = 0; kk < 8; ++kk) a[kk] = load_A(G + baseG, HH_, mt * 16, kk * 32, lane);
    #pragma unroll
    for (int nt = 0; nt < 8; ++nt) {
      v8f acc = {};
      #pragma unroll
      for (int kk = 0; kk < 8; ++kk)
        acc = wmma_f16(a[kk], load_B(wB, HH_, nt * 16, kk * 32, lane), acc);
      int n  = nt * 16 + (lane & 15);
      int mb = mt * 16 + 8 * (lane >> 4);
      float bb = b2[n];
      float resid[8];                            // batched residual loads
      #pragma unroll
      for (int r = 0; r < 8; ++r) resid[r] = X2out[baseX + (size_t)(mb + r) * PD_ + n];
      #pragma unroll
      for (int r = 0; r < 8; ++r)
        X2out[baseX + (size_t)(mb + r) * PD_ + n] = resid[r] + acc[r] + bb;
    }
  }
}

extern "C" void kernel_launch(void* const* d_in, const int* in_sizes, int n_in,
                              void* d_out, int out_size, void* d_ws, size_t ws_size,
                              hipStream_t stream) {
  (void)in_sizes; (void)n_in; (void)out_size; (void)ws_size;
  const float* seq   = (const float*)d_in[0];
  const float* pair  = (const float*)d_in[1];
  const float* sn_g  = (const float*)d_in[2];
  const float* sn_b  = (const float*)d_in[3];
  const float* Wq    = (const float*)d_in[4];
  const float* Wk    = (const float*)d_in[5];
  const float* Wyq   = (const float*)d_in[6];
  const float* Wyk   = (const float*)d_in[7];
  const float* Wdist = (const float*)d_in[8];
  const float* Wout  = (const float*)d_in[9];
  const float* bout  = (const float*)d_in[10];
  const float* rn_g  = (const float*)d_in[11];
  const float* rn_b  = (const float*)d_in[12];
  const float* Wrq   = (const float*)d_in[13];
  const float* Wrk   = (const float*)d_in[14];
  const float* Wrv   = (const float*)d_in[15];
  const float* brv   = (const float*)d_in[16];
  const float* Wro   = (const float*)d_in[17];
  const float* bro   = (const float*)d_in[18];
  const float* mn_g  = (const float*)d_in[19];
  const float* mn_b  = (const float*)d_in[20];
  const float* W1    = (const float*)d_in[21];
  const float* b1    = (const float*)d_in[22];
  const float* W2    = (const float*)d_in[23];
  const float* b2    = (const float*)d_in[24];

  char* ws = (char*)d_ws;
  const size_t XB = (size_t)B_ * P_ * P_ * PD_ * sizeof(float);
  float*    X   = (float*)ws;
  _Float16* G16 = (_Float16*)ws;                 // aliases X (X dead before G written)
  char* cur = ws + XB;
  float* a_side = (float*)cur;  cur += (size_t)B_ * P_ * PD_ * sizeof(float);
  float* b_side = (float*)cur;  cur += (size_t)B_ * P_ * PD_ * sizeof(float);
  _Float16* WoutT = (_Float16*)cur; cur += (size_t)PD_ * PD_ * 2;
  _Float16* WqkvT = (_Float16*)cur; cur += (size_t)3 * AD_ * PD_ * 2;
  _Float16* WroT  = (_Float16*)cur; cur += (size_t)PD_ * AD_ * 2;
  _Float16* W1T   = (_Float16*)cur; cur += (size_t)HH_ * PD_ * 2;
  _Float16* W2T   = (_Float16*)cur; cur += (size_t)PD_ * HH_ * 2;
  const size_t ATT = (size_t)B_ * P_ * P_ * AD_ * 2;
  _Float16* QA  = (_Float16*)cur; cur += ATT;
  _Float16* KA  = (_Float16*)cur; cur += ATT;
  _Float16* VA  = (_Float16*)cur; cur += ATT;
  _Float16* O16 = (_Float16*)cur; cur += ATT;
  float* X2 = (float*)d_out;

  dim3 gridPB(P_, B_);

  k_prep<<<dim3(B_ * P_), 256, 0, stream>>>(seq, sn_g, sn_b, Wq, Wk, Wyq, Wyk, a_side, b_side);

  k_cvtT<<<(PD_ * PD_ + 255) / 256, 256, 0, stream>>>(Wout, WoutT, PD_, PD_, 1.0f);
  k_cvtT<<<(PD_ * AD_ + 255) / 256, 256, 0, stream>>>(Wrq, WqkvT, PD_, AD_, 0.125f); // 1/sqrt(64)
  k_cvtT<<<(PD_ * AD_ + 255) / 256, 256, 0, stream>>>(Wrk, WqkvT + AD_ * PD_, PD_, AD_, 1.0f);
  k_cvtT<<<(PD_ * AD_ + 255) / 256, 256, 0, stream>>>(Wrv, WqkvT + 2 * AD_ * PD_, PD_, AD_, 1.0f);
  k_cvtT<<<(AD_ * PD_ + 255) / 256, 256, 0, stream>>>(Wro, WroT, AD_, PD_, 1.0f);
  k_cvtT<<<(PD_ * HH_ + 255) / 256, 256, 0, stream>>>(W1, W1T, PD_, HH_, 1.0f);
  k_cvtT<<<(HH_ * PD_ + 255) / 256, 256, 0, stream>>>(W2, W2T, HH_, PD_, 1.0f);

  k_pair<<<gridPB, 256, (P_ * PD_ + PD_ * PD_) * 2, stream>>>(
      a_side, b_side, Wdist, WoutT, bout, pair, X);

  k_lnqkv<<<gridPB, 256, (P_ * PD_ + 3 * AD_ * PD_) * 2, stream>>>(
      X, rn_g, rn_b, WqkvT, brv, QA, KA, VA);

  k_attn<<<gridPB, 256, (P_ * AD_ + AD_ * P_ + 8 * 16 * 32) * 2, stream>>>(QA, KA, VA, O16);

  k_attnout<<<gridPB, 256, (P_ * AD_ + PD_ * AD_) * 2, stream>>>(O16, WroT, bro, X, X2);

  k_mlp1<<<gridPB, 256, (P_ * PD_ + HH_ * PD_) * 2, stream>>>(X2, mn_g, mn_b, W1T, b1, G16);

  k_mlp2<<<gridPB, 256, (PD_ * HH_) * 2, stream>>>(G16, W2T, b2, X2);
}
